// pointnet2_seg_ssg_42760694399250
// MI455X (gfx1250) — matrix operand order repaired and checked by reading, not compiled, not run on hardware
//
#include <hip/hip_runtime.h>

// ============================================================================
// PointNet++ SSG segmentation forward, MI455X (gfx1250, wave32, WMMA).
//
// Roofline: ~30 GFLOP of matmul vs ~1 GB fp32 activation traffic -> memory
// bound at 23.3 TB/s. All channel-mixing einsums run through
// v_wmma_f32_16x16x32_f16 tiled GEMMs (f16 operands, f32 accumulate); fp32 is
// kept in HBM at layer boundaries because BatchNorm (training-mode stats)
// forces a global reduction between GEMM and activation anyway.
// Two GEMM variants: a fully-aligned double-buffered one (no bounds code,
// one barrier per k-step) for the dominant layers, and a general clamped
// variant for ragged Cin/Cout. B-tile transpose staging packs k-pairs into
// 32-bit LDS stores (8x ds_store_b32 per lane instead of 16x ds_store_b16).
// FPS is the latency-critical serial stage: one workgroup per batch keeps the
// 8192-entry running-distance array in LDS.
// ============================================================================

typedef __attribute__((ext_vector_type(16))) _Float16 v16h;
typedef __attribute__((ext_vector_type(8)))  float    v8f;
typedef __attribute__((ext_vector_type(4)))  _Float16 h4;
typedef __attribute__((ext_vector_type(2)))  _Float16 h2;

#define BATCH 16
#define NPTS  8192
#define NCLS  50

// ---------------------------------------------------------------------------
// Farthest point sampling: 1 block per batch, dists resident in LDS.
// Emits centroid index *before* the distance update (matches the lax.scan).
// ---------------------------------------------------------------------------
__global__ __launch_bounds__(256) void k_fps(const float* __restrict__ xyz,
                                             int n, int m, int* __restrict__ out) {
  const int b = blockIdx.x;
  __shared__ float sd[NPTS];
  __shared__ float rv[256];
  __shared__ int   ri[256];
  __shared__ int   s_cur;
  const float* p = xyz + (size_t)b * n * 3;
  for (int i = threadIdx.x; i < n; i += 256) sd[i] = 1e5f;
  if (threadIdx.x == 0) s_cur = 0;
  __syncthreads();
  for (int it = 0; it < m; ++it) {
    const int cur = s_cur;
    if (threadIdx.x == 0) out[b * m + it] = cur;
    const float cx = p[cur * 3 + 0], cy = p[cur * 3 + 1], cz = p[cur * 3 + 2];
    float bv = -1.0f; int bi = 0;
    for (int i = threadIdx.x; i < n; i += 256) {
      const float dx = p[i * 3 + 0] - cx;
      const float dy = p[i * 3 + 1] - cy;
      const float dz = p[i * 3 + 2] - cz;
      float d2 = dx * dx + dy * dy + dz * dz;
      if (d2 < 1e-7f) d2 = 1e-7f;
      const float nd = fminf(sd[i], sqrtf(d2));
      sd[i] = nd;
      if (nd > bv) { bv = nd; bi = i; }
    }
    rv[threadIdx.x] = bv; ri[threadIdx.x] = bi;
    __syncthreads();
    for (int s = 128; s > 0; s >>= 1) {
      if (threadIdx.x < s) {
        const float ov = rv[threadIdx.x + s]; const int oi = ri[threadIdx.x + s];
        if (ov > rv[threadIdx.x] || (ov == rv[threadIdx.x] && oi < ri[threadIdx.x])) {
          rv[threadIdx.x] = ov; ri[threadIdx.x] = oi;
        }
      }
      __syncthreads();
    }
    if (threadIdx.x == 0) s_cur = ri[0];
    __syncthreads();
  }
}

// gather xyz rows by index: out[i] = xyz[b, idx[i]]
__global__ void k_gather_xyz(const float* __restrict__ xyz, const int* __restrict__ idx,
                             float* __restrict__ out, int n, int m, int total) {
  const int i = blockIdx.x * blockDim.x + threadIdx.x;
  if (i >= total) return;
  const int b = i / m;
  const int j = idx[i];
  const float* s = xyz + ((size_t)b * n + j) * 3;
  out[i * 3 + 0] = s[0]; out[i * 3 + 1] = s[1]; out[i * 3 + 2] = s[2];
}

// ---------------------------------------------------------------------------
// Ball query: first K indices (ascending) within radius, padded with first hit.
// ---------------------------------------------------------------------------
__global__ void k_ball_query(const float* __restrict__ xyz, const float* __restrict__ nxyz,
                             int n, int m, float radius, int K,
                             int* __restrict__ out, int total) {
  const int i = blockIdx.x * blockDim.x + threadIdx.x;  // over B*m
  if (i >= total) return;
  const int b = i / m;
  const float* p = xyz + (size_t)b * n * 3;
  const float cx = nxyz[i * 3 + 0], cy = nxyz[i * 3 + 1], cz = nxyz[i * 3 + 2];
  int cnt = 0, first = -1;
  for (int j = 0; j < n && cnt < K; ++j) {
    const float dx = p[j * 3 + 0] - cx;
    const float dy = p[j * 3 + 1] - cy;
    const float dz = p[j * 3 + 2] - cz;
    float d2 = dx * dx + dy * dy + dz * dz;
    if (d2 < 1e-7f) d2 = 1e-7f;
    if (sqrtf(d2) <= radius) {
      if (first < 0) first = j;
      out[(size_t)i * K + cnt] = j;
      ++cnt;
    }
  }
  const int pad = (first < 0) ? (n - 1) : first;
  for (; cnt < K; ++cnt) out[(size_t)i * K + cnt] = pad;
}

// group (element-parallel, coalesced): out[b,m,k,c] = c<3 ? xyz[gi]-center : feat[gi]
__global__ void k_group(const float* __restrict__ xyz, const float* __restrict__ feat, int Cf,
                        const float* __restrict__ nxyz, const int* __restrict__ gi,
                        float* __restrict__ out, int n, int m, int K, int total /*B*m*K*C*/) {
  const int i = blockIdx.x * blockDim.x + threadIdx.x;
  if (i >= total) return;
  const int C = 3 + Cf;
  const int c = i % C;
  const int bmk = i / C;
  const int b = bmk / (m * K);
  const int j = gi[bmk];
  float v;
  if (c < 3) v = xyz[((size_t)b * n + j) * 3 + c] - nxyz[(size_t)(bmk / K) * 3 + c];
  else       v = feat[((size_t)b * n + j) * Cf + (c - 3)];
  out[i] = v;
}

// ---------------------------------------------------------------------------
// WMMA GEMM, aligned fast variant: R%16==0, Cin%32==0, Cout%64==0.
// Double-buffered LDS tiles, no bounds code, one barrier per k-step.
// 4 waves/block share the A-tile; each wave owns a transposed B-tile and a
// 16x16 output tile. A/B fragments use contiguous K (k = e + 16*hsel).
// ---------------------------------------------------------------------------
#define GEMM_NW 4
__global__ __launch_bounds__(32 * GEMM_NW) void k_gemm_f16_al(
    const float* __restrict__ X, const float* __restrict__ W,
    float* __restrict__ Y, int R, int Cin, int Cout) {
  __shared__ _Float16 As[2][16][32];
  __shared__ _Float16 Bs[2][GEMM_NW][16][32];  // transposed: [n][k]
  const int tid = threadIdx.x;
  const int wave = tid >> 5;
  const int lane = tid & 31;
  const int rowBase = blockIdx.x * 16;
  const int colBase = blockIdx.y * (16 * GEMM_NW) + wave * 16;
  const int mrow = lane & 15;   // A row / B column / D column
  const int hsel = lane >> 4;   // half select
  const int ar  = tid >> 3;        // A cooperative-load row (0..15)
  const int ak4 = (tid & 7) * 4;   // A k-group of 4
  const int bk2 = (lane & 15) * 2; // B k-pair base
  const int bnb = (lane >> 4) * 8; // B n base (0 or 8)

  auto loadA = [&](int k0, int bsel) {
    const float* src = X + (size_t)(rowBase + ar) * Cin + k0 + ak4;
    h4 hv = {(_Float16)src[0], (_Float16)src[1], (_Float16)src[2], (_Float16)src[3]};
    *(h4*)&As[bsel][ar][ak4] = hv;
  };
  // lane owns k-pair {bk2,bk2+1} x 8 columns -> 8 packed 32-bit LDS stores
  auto loadB = [&](int k0, int bsel) {
    const float* s0 = W + (size_t)(k0 + bk2) * Cout + colBase + bnb;
    const float* s1 = s0 + Cout;
#pragma unroll
    for (int nn = 0; nn < 8; ++nn) {
      h2 hv = {(_Float16)s0[nn], (_Float16)s1[nn]};
      *(h2*)&Bs[bsel][wave][bnb + nn][bk2] = hv;
    }
  };

  v8f acc = {};
  loadA(0, 0);
  loadB(0, 0);
  __syncthreads();
  int bsel = 0;
  for (int k0 = 0; k0 < Cin; k0 += 32) {
    // stage next tile into the other buffer while this one computes
    if (k0 + 32 < Cin) { loadA(k0 + 32, bsel ^ 1); loadB(k0 + 32, bsel ^ 1); }
    v16h a, bf;
    // A fragment: lane holds row mrow; K = e + 16*hsel (contiguous)
#pragma unroll
    for (int e = 0; e < 16; e += 2) {
      const int vi = e >> 1;
      const int ka = 2 * vi + 16 * hsel;
      a[e]     = As[bsel][mrow][ka];
      a[e + 1] = As[bsel][mrow][ka + 1];
    }
    // B fragment: lane holds column mrow; K = e + 16*hsel (contiguous in LDS)
#pragma unroll
    for (int e = 0; e < 16; ++e) bf[e] = Bs[bsel][wave][mrow][e + 16 * hsel];
    acc = __builtin_amdgcn_wmma_f32_16x16x32_f16(false, a, false, bf,
                                                 (short)0, acc, false, false);
    __syncthreads();
    bsel ^= 1;
  }
  // D layout: VGPR r -> row r + 8*hsel, column mrow
#pragma unroll
  for (int r = 0; r < 8; ++r)
    Y[(size_t)(rowBase + r + 8 * hsel) * Cout + colBase + mrow] = acc[r];
}

// ---------------------------------------------------------------------------
// WMMA GEMM, general variant (ragged Cin/Cout): branchless clamped staging.
// ---------------------------------------------------------------------------
__global__ __launch_bounds__(32 * GEMM_NW) void k_gemm_f16(
    const float* __restrict__ X, const float* __restrict__ W,
    float* __restrict__ Y, int R, int Cin, int Cout) {
  __shared__ _Float16 As[16][32];
  __shared__ _Float16 Bs[GEMM_NW][16][32];  // transposed: [n][k]
  const int tid = threadIdx.x;
  const int wave = tid >> 5;
  const int lane = tid & 31;
  const int rowBase = blockIdx.x * 16;
  const int colBase = blockIdx.y * (16 * GEMM_NW) + wave * 16;
  const int mrow = lane & 15;
  const int hsel = lane >> 4;
  const int ar  = tid >> 3;
  const int ak4 = (tid & 7) * 4;
  const int bk2 = (lane & 15) * 2;
  const int bnb = (lane >> 4) * 8;
  const bool rowsFull = (rowBase + 16 <= R);
  const bool colsFull = (colBase + 16 <= Cout);
  v8f acc = {};
  for (int k0 = 0; k0 < Cin; k0 += 32) {
    const bool kFull = (k0 + 32 <= Cin);
    // ---- A tile ----
    if (rowsFull && kFull) {
      const float* src = X + (size_t)(rowBase + ar) * Cin + k0 + ak4;
      h4 hv = {(_Float16)src[0], (_Float16)src[1], (_Float16)src[2], (_Float16)src[3]};
      *(h4*)&As[ar][ak4] = hv;
    } else {
      const int gr = rowBase + ar;
      const int rr = gr < R ? gr : R - 1;
      const float* src = X + (size_t)rr * Cin;
      h4 hv;
#pragma unroll
      for (int e = 0; e < 4; ++e) {
        const int gk = k0 + ak4 + e;
        const int kk = gk < Cin ? gk : Cin - 1;
        const float v = src[kk];
        hv[e] = (_Float16)((gr < R && gk < Cin) ? v : 0.0f);
      }
      *(h4*)&As[ar][ak4] = hv;
    }
    // ---- B tile: lane owns k-pair x 8 columns, packed 32-bit LDS stores ----
    if (kFull && colsFull) {
      const float* s0 = W + (size_t)(k0 + bk2) * Cout + colBase + bnb;
      const float* s1 = s0 + Cout;
#pragma unroll
      for (int nn = 0; nn < 8; ++nn) {
        h2 hv = {(_Float16)s0[nn], (_Float16)s1[nn]};
        *(h2*)&Bs[wave][bnb + nn][bk2] = hv;
      }
    } else {
      const int gk0 = k0 + bk2, gk1 = gk0 + 1;
      const int kk0 = gk0 < Cin ? gk0 : Cin - 1;
      const int kk1 = gk1 < Cin ? gk1 : Cin - 1;
      const float* s0 = W + (size_t)kk0 * Cout;
      const float* s1 = W + (size_t)kk1 * Cout;
#pragma unroll
      for (int nn = 0; nn < 8; ++nn) {
        const int gc = colBase + bnb + nn;
        const int cc = gc < Cout ? gc : Cout - 1;
        const float v0 = s0[cc];
        const float v1 = s1[cc];
        const bool okc = gc < Cout;
        h2 hv = {(_Float16)((gk0 < Cin && okc) ? v0 : 0.0f),
                 (_Float16)((gk1 < Cin && okc) ? v1 : 0.0f)};
        *(h2*)&Bs[wave][bnb + nn][bk2] = hv;
      }
    }
    __syncthreads();
    v16h a, bf;
#pragma unroll
    for (int e = 0; e < 16; e += 2) {
      const int vi = e >> 1;
      const int ka = 2 * vi + 16 * hsel;
      a[e]     = As[mrow][ka];
      a[e + 1] = As[mrow][ka + 1];
    }
#pragma unroll
    for (int e = 0; e < 16; ++e) bf[e] = Bs[wave][mrow][e + 16 * hsel];
    acc = __builtin_amdgcn_wmma_f32_16x16x32_f16(false, a, false, bf,
                                                 (short)0, acc, false, false);
    __syncthreads();
  }
  if (rowsFull && colsFull) {
#pragma unroll
    for (int r = 0; r < 8; ++r)
      Y[(size_t)(rowBase + r + 8 * hsel) * Cout + colBase + mrow] = acc[r];
  } else {
#pragma unroll
    for (int r = 0; r < 8; ++r) {
      const int gr = rowBase + r + 8 * hsel;
      const int gc = colBase + mrow;
      if (gr < R && gc < Cout) Y[(size_t)gr * Cout + gc] = acc[r];
    }
  }
}

// per-channel sum / sumsq (one block per channel; no atomics)
__global__ __launch_bounds__(256) void k_colstat(const float* __restrict__ X,
                                                 float* __restrict__ stats, int R, int C) {
  const int c = blockIdx.x;
  __shared__ float s1[256], s2[256];
  float a = 0.0f, b = 0.0f;
  for (int r = threadIdx.x; r < R; r += 256) {
    const float v = X[(size_t)r * C + c];
    a += v; b += v * v;
  }
  s1[threadIdx.x] = a; s2[threadIdx.x] = b;
  __syncthreads();
  for (int s = 128; s > 0; s >>= 1) {
    if (threadIdx.x < s) { s1[threadIdx.x] += s1[threadIdx.x + s]; s2[threadIdx.x] += s2[threadIdx.x + s]; }
    __syncthreads();
  }
  if (threadIdx.x == 0) { stats[2 * c] = s1[0]; stats[2 * c + 1] = s2[0]; }
}

// BN (training-mode biased stats) + ReLU, in place
__global__ void k_bn_relu(float* __restrict__ X, const float* __restrict__ stats,
                          const float* __restrict__ g, const float* __restrict__ bb,
                          int R, int C) {
  const size_t i = (size_t)blockIdx.x * blockDim.x + threadIdx.x;
  if (i >= (size_t)R * C) return;
  const int c = (int)(i % C);
  const float mu = stats[2 * c] / (float)R;
  const float var = stats[2 * c + 1] / (float)R - mu * mu;
  const float v = (X[i] - mu) * rsqrtf(var + 1e-5f) * g[c] + bb[c];
  X[i] = v > 0.0f ? v : 0.0f;
}

// max over the K (neighbor) axis: x is (BM, K, C) -> out (BM, C)
__global__ void k_maxpool(const float* __restrict__ x, float* __restrict__ out,
                          int K, int C, int total) {
  const int i = blockIdx.x * blockDim.x + threadIdx.x;
  if (i >= total) return;
  const int c = i % C;
  const int bm = i / C;
  const float* p = x + (size_t)bm * K * C + c;
  float v = p[0];
  for (int k = 1; k < K; ++k) v = fmaxf(v, p[(size_t)k * C]);
  out[i] = v;
}

// SA3 input (element-parallel): [l2_xyz(3), l2_points(256)]
__global__ void k_sa3_concat(const float* __restrict__ xyz, const float* __restrict__ pts,
                             float* __restrict__ out, int total /*rows*259*/) {
  const int i = blockIdx.x * blockDim.x + threadIdx.x;
  if (i >= total) return;
  const int c = i % 259;
  const int r = i / 259;
  out[i] = (c < 3) ? xyz[r * 3 + c] : pts[(size_t)r * 256 + (c - 3)];
}

// FP1 input (element-parallel): [broadcast l3(1024), l2_points(256)]
__global__ void k_fp1_concat(const float* __restrict__ l3, const float* __restrict__ l2p,
                             float* __restrict__ out, int total /*rows*1280*/) {
  const int i = blockIdx.x * blockDim.x + threadIdx.x;
  if (i >= total) return;
  const int c = i % 1280;
  const int r = i / 1280;
  const int b = r / 128;
  out[i] = (c < 1024) ? l3[(size_t)b * 1024 + c] : l2p[(size_t)r * 256 + (c - 1024)];
}

// generic 2-way channel concat (element-parallel)
__global__ void k_concat2(const float* __restrict__ a, int C1,
                          const float* __restrict__ b, int C2,
                          float* __restrict__ out, int total /*rows*(C1+C2)*/) {
  const int i = blockIdx.x * blockDim.x + threadIdx.x;
  if (i >= total) return;
  const int C = C1 + C2;
  const int c = i % C;
  const int r = i / C;
  out[i] = (c < C1) ? a[(size_t)r * C1 + c] : b[(size_t)r * C2 + (c - C1)];
}

// FP3 input (element-parallel): [interp(128), l0_points(3), l0_xyz(3)]
__global__ void k_fp3_concat(const float* __restrict__ interp, const float* __restrict__ l0p,
                             const float* __restrict__ l0x, float* __restrict__ out,
                             int total /*rows*134*/) {
  const int i = blockIdx.x * blockDim.x + threadIdx.x;
  if (i >= total) return;
  const int c = i % 134;
  const int r = i / 134;
  float v;
  if (c < 128)      v = interp[(size_t)r * 128 + c];
  else if (c < 131) v = l0p[r * 3 + (c - 128)];
  else              v = l0x[r * 3 + (c - 131)];
  out[i] = v;
}

// 3-NN search: per query point, write 3 indices + normalized inverse-dist weights
__global__ void k_knn3(const float* __restrict__ xyz1, const float* __restrict__ xyz2,
                       int n1, int n2, int* __restrict__ idx, float* __restrict__ wts,
                       int total /*B*n1*/) {
  const int i = blockIdx.x * blockDim.x + threadIdx.x;
  if (i >= total) return;
  const int b = i / n1;
  const float px = xyz1[i * 3 + 0], py = xyz1[i * 3 + 1], pz = xyz1[i * 3 + 2];
  const float* q = xyz2 + (size_t)b * n2 * 3;
  float d0 = 1e30f, d1 = 1e30f, d2 = 1e30f;
  int i0 = 0, i1 = 0, i2 = 0;
  for (int j = 0; j < n2; ++j) {
    const float dx = q[j * 3 + 0] - px;
    const float dy = q[j * 3 + 1] - py;
    const float dz = q[j * 3 + 2] - pz;
    float dd = dx * dx + dy * dy + dz * dz;
    if (dd < 1e-7f) dd = 1e-7f;
    const float d = sqrtf(dd);
    if (d < d0)      { d2 = d1; i2 = i1; d1 = d0; i1 = i0; d0 = d; i0 = j; }
    else if (d < d1) { d2 = d1; i2 = i1; d1 = d;  i1 = j; }
    else if (d < d2) { d2 = d;  i2 = j; }
  }
  float w0 = 1.0f / (d0 + 1e-8f), w1 = 1.0f / (d1 + 1e-8f), w2 = 1.0f / (d2 + 1e-8f);
  const float s = w0 + w1 + w2;
  idx[3 * i + 0] = i0; idx[3 * i + 1] = i1; idx[3 * i + 2] = i2;
  wts[3 * i + 0] = w0 / s; wts[3 * i + 1] = w1 / s; wts[3 * i + 2] = w2 / s;
}

// apply 3-NN interpolation (element-parallel, coalesced over channels)
__global__ void k_interp_apply(const float* __restrict__ pts2, const int* __restrict__ idx,
                               const float* __restrict__ wts, float* __restrict__ out,
                               int n1, int n2, int C, int total /*B*n1*C*/) {
  const int i = blockIdx.x * blockDim.x + threadIdx.x;
  if (i >= total) return;
  const int c = i % C;
  const int p = i / C;
  const int b = p / n1;
  const float* f = pts2 + (size_t)b * n2 * C;
  out[i] = wts[3 * p + 0] * f[(size_t)idx[3 * p + 0] * C + c]
         + wts[3 * p + 1] * f[(size_t)idx[3 * p + 1] * C + c]
         + wts[3 * p + 2] * f[(size_t)idx[3 * p + 2] * C + c];
}

// logits (B*N, 50) -> out (B, 50, N) with bias
__global__ void k_bias_transpose(const float* __restrict__ logits, const float* __restrict__ bias,
                                 float* __restrict__ out, int total) {
  const int i = blockIdx.x * blockDim.x + threadIdx.x;
  if (i >= total) return;
  const int n = i % NPTS;
  const int bc = i / NPTS;
  const int c = bc % NCLS;
  const int b = bc / NCLS;
  out[i] = logits[((size_t)b * NPTS + n) * NCLS + c] + bias[c];
}

// ============================================================================
// Host launcher
// ============================================================================
extern "C" void kernel_launch(void* const* d_in, const int* in_sizes, int n_in,
                              void* d_out, int out_size, void* d_ws, size_t ws_size,
                              hipStream_t stream) {
  (void)in_sizes; (void)n_in; (void)out_size; (void)ws_size;
  // --- input order: top-level dict insertion order (l0_xyz, l0_points, params),
  //     params flattened as a pytree (dict keys sorted; mlp dicts b,g,w). ---
  const float* l0_xyz = (const float*)d_in[0];
  const float* l0_pts = (const float*)d_in[1];
  const float* bn1_b  = (const float*)d_in[2];
  const float* bn1_g  = (const float*)d_in[3];
  const float* cls_b  = (const float*)d_in[4];
  const float* cls_w  = (const float*)d_in[5];
  const float* conv1w = (const float*)d_in[6];
  const float *fp1_b[2] = {(const float*)d_in[7], (const float*)d_in[8]};
  const float *fp1_g[2] = {(const float*)d_in[9], (const float*)d_in[10]};
  const float *fp1_w[2] = {(const float*)d_in[11], (const float*)d_in[12]};
  const float *fp2_b[2] = {(const float*)d_in[13], (const float*)d_in[14]};
  const float *fp2_g[2] = {(const float*)d_in[15], (const float*)d_in[16]};
  const float *fp2_w[2] = {(const float*)d_in[17], (const float*)d_in[18]};
  const float *fp3_b[3] = {(const float*)d_in[19], (const float*)d_in[20], (const float*)d_in[21]};
  const float *fp3_g[3] = {(const float*)d_in[22], (const float*)d_in[23], (const float*)d_in[24]};
  const float *fp3_w[3] = {(const float*)d_in[25], (const float*)d_in[26], (const float*)d_in[27]};
  const float *sa1_b[3] = {(const float*)d_in[28], (const float*)d_in[29], (const float*)d_in[30]};
  const float *sa1_g[3] = {(const float*)d_in[31], (const float*)d_in[32], (const float*)d_in[33]};
  const float *sa1_w[3] = {(const float*)d_in[34], (const float*)d_in[35], (const float*)d_in[36]};
  const float *sa2_b[3] = {(const float*)d_in[37], (const float*)d_in[38], (const float*)d_in[39]};
  const float *sa2_g[3] = {(const float*)d_in[40], (const float*)d_in[41], (const float*)d_in[42]};
  const float *sa2_w[3] = {(const float*)d_in[43], (const float*)d_in[44], (const float*)d_in[45]};
  const float *sa3_b[3] = {(const float*)d_in[46], (const float*)d_in[47], (const float*)d_in[48]};
  const float *sa3_g[3] = {(const float*)d_in[49], (const float*)d_in[50], (const float*)d_in[51]};
  const float *sa3_w[3] = {(const float*)d_in[52], (const float*)d_in[53], (const float*)d_in[54]};

  // --- workspace bump allocator ---
  char* wp = (char*)d_ws;
  auto alloc = [&](size_t bytes) -> void* {
    void* p = (void*)wp;
    wp += (bytes + 255) & ~(size_t)255;
    return p;
  };
  const size_t PPF = (size_t)BATCH * 512 * 32 * 128;  // 33.5M floats, max activation
  float* pp0 = (float*)alloc(PPF * 4);
  float* pp1 = (float*)alloc(PPF * 4);
  float* stats = (float*)alloc(2048 * 4);
  int*   fps1 = (int*)alloc((size_t)BATCH * 512 * 4);
  float* l1_xyz = (float*)alloc((size_t)BATCH * 512 * 3 * 4);
  int*   bq1 = (int*)alloc((size_t)BATCH * 512 * 32 * 4);
  float* l1_pts = (float*)alloc((size_t)BATCH * 512 * 128 * 4);
  int*   fps2 = (int*)alloc((size_t)BATCH * 128 * 4);
  float* l2_xyz = (float*)alloc((size_t)BATCH * 128 * 3 * 4);
  int*   bq2 = (int*)alloc((size_t)BATCH * 128 * 64 * 4);
  float* l2_pts = (float*)alloc((size_t)BATCH * 128 * 256 * 4);
  float* l3_pts = (float*)alloc((size_t)BATCH * 1024 * 4);
  float* l2p_new = (float*)alloc((size_t)BATCH * 128 * 256 * 4);
  float* interp2 = (float*)alloc((size_t)BATCH * 512 * 256 * 4);
  float* l1p_new = (float*)alloc((size_t)BATCH * 512 * 128 * 4);
  float* interp3 = (float*)alloc((size_t)BATCH * NPTS * 128 * 4);  // reused for logits
  int*   knn_i = (int*)alloc((size_t)BATCH * NPTS * 3 * 4);
  float* knn_w = (float*)alloc((size_t)BATCH * NPTS * 3 * 4);

  auto cdiv = [](int a, int b) { return (a + b - 1) / b; };
  auto gemm = [&](const float* Xi, const float* Wt, float* Yo, int R, int Cin, int Cout) {
    dim3 grid(cdiv(R, 16), cdiv(Cout, 16 * GEMM_NW));
    if ((R & 15) == 0 && (Cin & 31) == 0 && (Cout & 63) == 0)
      k_gemm_f16_al<<<grid, 32 * GEMM_NW, 0, stream>>>(Xi, Wt, Yo, R, Cin, Cout);
    else
      k_gemm_f16<<<grid, 32 * GEMM_NW, 0, stream>>>(Xi, Wt, Yo, R, Cin, Cout);
  };
  auto layer = [&](const float* Xi, float* Yo, const float* Wt,
                   const float* g, const float* bb, int R, int Cin, int Cout) {
    gemm(Xi, Wt, Yo, R, Cin, Cout);
    k_colstat<<<Cout, 256, 0, stream>>>(Yo, stats, R, Cout);
    const size_t tot = (size_t)R * Cout;
    k_bn_relu<<<(unsigned)((tot + 255) / 256), 256, 0, stream>>>(Yo, stats, g, bb, R, Cout);
  };

  // ================= SA1: 8192 -> 512 centroids, K=32, 6->64->64->128 =======
  {
    const int M = 512, K = 32, n = NPTS;
    k_fps<<<BATCH, 256, 0, stream>>>(l0_xyz, n, M, fps1);
    const int t = BATCH * M;
    k_gather_xyz<<<cdiv(t, 256), 256, 0, stream>>>(l0_xyz, fps1, l1_xyz, n, M, t);
    k_ball_query<<<cdiv(t, 256), 256, 0, stream>>>(l0_xyz, l1_xyz, n, M, 0.2f, K, bq1, t);
    const int R = BATCH * M * K;
    const int te = R * 6;
    k_group<<<cdiv(te, 256), 256, 0, stream>>>(l0_xyz, l0_pts, 3, l1_xyz, bq1, pp0, n, M, K, te);
    layer(pp0, pp1, sa1_w[0], sa1_g[0], sa1_b[0], R, 6, 64);
    layer(pp1, pp0, sa1_w[1], sa1_g[1], sa1_b[1], R, 64, 64);
    layer(pp0, pp1, sa1_w[2], sa1_g[2], sa1_b[2], R, 64, 128);
    const int tp = BATCH * M * 128;
    k_maxpool<<<cdiv(tp, 256), 256, 0, stream>>>(pp1, l1_pts, K, 128, tp);
  }
  // ================= SA2: 512 -> 128, K=64, 131->128->128->256 ==============
  {
    const int M = 128, K = 64, n = 512;
    k_fps<<<BATCH, 256, 0, stream>>>(l1_xyz, n, M, fps2);
    const int t = BATCH * M;
    k_gather_xyz<<<cdiv(t, 256), 256, 0, stream>>>(l1_xyz, fps2, l2_xyz, n, M, t);
    k_ball_query<<<cdiv(t, 256), 256, 0, stream>>>(l1_xyz, l2_xyz, n, M, 0.4f, K, bq2, t);
    const int R = BATCH * M * K;
    const int te = R * 131;
    k_group<<<cdiv(te, 256), 256, 0, stream>>>(l1_xyz, l1_pts, 128, l2_xyz, bq2, pp0, n, M, K, te);
    layer(pp0, pp1, sa2_w[0], sa2_g[0], sa2_b[0], R, 131, 128);
    layer(pp1, pp0, sa2_w[1], sa2_g[1], sa2_b[1], R, 128, 128);
    layer(pp0, pp1, sa2_w[2], sa2_g[2], sa2_b[2], R, 128, 256);
    const int tp = BATCH * M * 256;
    k_maxpool<<<cdiv(tp, 256), 256, 0, stream>>>(pp1, l2_pts, K, 256, tp);
  }
  // ================= SA3 (group-all): 259->256->512->1024, max over 128 =====
  {
    const int R = BATCH * 128;
    const int te = R * 259;
    k_sa3_concat<<<cdiv(te, 256), 256, 0, stream>>>(l2_xyz, l2_pts, pp0, te);
    layer(pp0, pp1, sa3_w[0], sa3_g[0], sa3_b[0], R, 259, 256);
    layer(pp1, pp0, sa3_w[1], sa3_g[1], sa3_b[1], R, 256, 512);
    layer(pp0, pp1, sa3_w[2], sa3_g[2], sa3_b[2], R, 512, 1024);
    const int tp = BATCH * 1024;
    k_maxpool<<<cdiv(tp, 256), 256, 0, stream>>>(pp1, l3_pts, 128, 1024, tp);
  }
  // ================= FP1: broadcast l3 + l2_points -> 1280->256->256 ========
  {
    const int R = BATCH * 128;
    const int te = R * 1280;
    k_fp1_concat<<<cdiv(te, 256), 256, 0, stream>>>(l3_pts, l2_pts, pp0, te);
    layer(pp0, pp1, fp1_w[0], fp1_g[0], fp1_b[0], R, 1280, 256);
    layer(pp1, l2p_new, fp1_w[1], fp1_g[1], fp1_b[1], R, 256, 256);
  }
  // ================= FP2: interp(128->512) + l1_points -> 384->256->128 =====
  {
    const int R = BATCH * 512;
    k_knn3<<<cdiv(R, 256), 256, 0, stream>>>(l1_xyz, l2_xyz, 512, 128, knn_i, knn_w, R);
    const int ti = R * 256;
    k_interp_apply<<<cdiv(ti, 256), 256, 0, stream>>>(l2p_new, knn_i, knn_w, interp2,
                                                      512, 128, 256, ti);
    const int te = R * 384;
    k_concat2<<<cdiv(te, 256), 256, 0, stream>>>(interp2, 256, l1_pts, 128, pp0, te);
    layer(pp0, pp1, fp2_w[0], fp2_g[0], fp2_b[0], R, 384, 256);
    layer(pp1, l1p_new, fp2_w[1], fp2_g[1], fp2_b[1], R, 256, 128);
  }
  // ================= FP3: interp(512->8192) + [l0_pts,l0_xyz] -> 134->128^3 ==
  {
    const int R = BATCH * NPTS;
    k_knn3<<<cdiv(R, 256), 256, 0, stream>>>(l0_xyz, l1_xyz, NPTS, 512, knn_i, knn_w, R);
    const int ti = R * 128;
    k_interp_apply<<<cdiv(ti, 256), 256, 0, stream>>>(l1p_new, knn_i, knn_w, interp3,
                                                      NPTS, 512, 128, ti);
    const int te = R * 134;
    k_fp3_concat<<<cdiv(te, 256), 256, 0, stream>>>(interp3, l0_pts, l0_xyz, pp0, te);
    layer(pp0, pp1, fp3_w[0], fp3_g[0], fp3_b[0], R, 134, 128);
    layer(pp1, pp0, fp3_w[1], fp3_g[1], fp3_b[1], R, 128, 128);
    layer(pp0, pp1, fp3_w[2], fp3_g[2], fp3_b[2], R, 128, 128);
  }
  // ================= head: conv1 (BN+ReLU) then classifier ==================
  {
    const int R = BATCH * NPTS;
    layer(pp1, pp0, conv1w, bn1_g, bn1_b, R, 128, 128);
    float* logits = interp3;  // free by now, 6.5M floats needed
    gemm(pp0, cls_w, logits, R, 128, NCLS);
    const int t = BATCH * NCLS * NPTS;
    k_bias_transpose<<<cdiv(t, 256), 256, 0, stream>>>(logits, cls_b, (float*)d_out, t);
  }
}
